// VoxDetHead_65111704207888
// MI455X (gfx1250) — matrix operand order repaired and checked by reading, not compiled, not run on hardware
//
#include <hip/hip_runtime.h>
#include <hip/hip_bf16.h>

typedef __attribute__((ext_vector_type(16))) _Float16 v16h;
typedef __attribute__((ext_vector_type(8)))  float    v8f;

// ---- problem dims (fixed by the reference) ----
#define XF 96
#define YF 96
#define ZF 16
#define NVOX (XF*YF*ZF)      // 147456
#define XD 48
#define YD 48
#define ZD 8
#define ND (XD*YD*ZD)        // 18432
#define CIN 64
#define MID 32
#define NBOX 6
#define NCLS 18
#define NLAYER 2

// =====================================================================
// conv1x1 as WMMA GEMM: out[O][N] = W[O][K] * in[K][N]   (batched via blockIdx.z)
// one wave (32 lanes) per 16-voxel column strip; f32->f16 convert in regs,
// f32 accumulate via v_wmma_f32_16x16x32_f16.
// Full M-tiles (O multiple-of-16 path) take a wave-uniform branch with
// unmasked A loads and unguarded strided stores; only ragged tail tiles
// (O=6 / O=18) pay for masking.
// =====================================================================
__global__ __launch_bounds__(256)
void k_conv1x1_wmma(const float* __restrict__ in, const float* __restrict__ w,
                    float* __restrict__ out, int O, int K, int N,
                    long inBStride, long outBStride)
{
    const float* inp  = in  + (long)blockIdx.z * inBStride;
    float*       outp = out + (long)blockIdx.z * outBStride;
    int lane  = threadIdx.x & 31;
    int wave  = threadIdx.x >> 5;
    int ntile = blockIdx.x * 8 + wave;
    int n0    = ntile * 16;
    if (n0 >= N) return;                       // wave-uniform
    int  col = lane & 15;
    bool hi  = lane >= 16;
    int  Mt  = (O + 15) >> 4;
    for (int mt = 0; mt < Mt; ++mt) {
        bool  full = (mt * 16 + 16) <= O;      // wave-uniform
        int   m    = mt * 16 + col;
        int   mc   = (m < O) ? m : (O - 1);    // clamped row (always valid)
        float msk  = (m < O) ? 1.f : 0.f;      // zero-out padding rows
        const float* wrow = w + (long)mc * K;
        v8f acc = {};
        for (int kb = 0; kb < K; kb += 32) {
            v16h am, bm;
            if (full) {
                #pragma unroll
                for (int h = 0; h < 16; ++h) { // ISA 16-bit A 16x32 layout
                    int ka = kb + h + ((h >= 8) ? 8 : 0) + (hi ? 8 : 0);
                    am[h] = (_Float16)wrow[ka];
                }
            } else {
                #pragma unroll
                for (int h = 0; h < 16; ++h) {
                    int ka = kb + h + ((h >= 8) ? 8 : 0) + (hi ? 8 : 0);
                    am[h] = (_Float16)(wrow[ka] * msk);
                }
            }
            #pragma unroll
            for (int h = 0; h < 16; ++h) {     // B 32x16: same K order as A (table layout)
                int kk = kb + h + ((h >= 8) ? 8 : 0) + (hi ? 8 : 0);
                bm[h] = (_Float16)inp[(long)kk * N + n0 + col];
            }
            acc = __builtin_amdgcn_wmma_f32_16x16x32_f16(
                      false, am, false, bm, (short)0, acc, false, false);
        }
        if (full) {                            // unguarded coalesced stores
            float* po = outp + (long)(mt * 16 + (hi ? 8 : 0)) * N + n0 + col;
            #pragma unroll
            for (int r = 0; r < 8; ++r) po[(long)r * N] = acc[r];
        } else {
            #pragma unroll
            for (int r = 0; r < 8; ++r) {      // C/D f32 16x16 layout
                int mrow = mt * 16 + r + (hi ? 8 : 0);
                if (mrow < O) outp[(long)mrow * N + n0 + col] = acc[r];
            }
        }
    }
}

// =====================================================================
// conv3x3x3 (SAME, no bias) as implicit GEMM: O=32, Cin=64, K=1728.
// B loader decodes (ic,dx,dy,dz); borders handled by clamped-address load
// + boundary-mask multiply (no divergent loads).
// =====================================================================
__global__ __launch_bounds__(256)
void k_conv3x3_wmma(const float* __restrict__ in, const float* __restrict__ w,
                    float* __restrict__ out)
{
    const int K = CIN * 27;                    // 1728
    int lane  = threadIdx.x & 31;
    int wave  = threadIdx.x >> 5;
    int ntile = blockIdx.x * 8 + wave;
    int n0    = ntile * 16;
    int  col = lane & 15;
    bool hi  = lane >= 16;
    int v = n0 + col;
    int z = v & (ZF - 1);
    int y = (v / ZF) % YF;
    int x = v / (ZF * YF);
    const float* wrow0 = w + (long)col * K;
    const float* wrow1 = w + (long)(16 + col) * K;
    v8f acc0 = {}, acc1 = {};
    for (int kb = 0; kb < K; kb += 32) {
        v16h bm;
        #pragma unroll
        for (int h = 0; h < 16; ++h) {
            int k  = kb + h + ((h >= 8) ? 8 : 0) + (hi ? 8 : 0);
            int ic = k / 27;
            int r  = k - ic * 27;
            int q9 = r / 9;
            int dx = q9 - 1;
            int ry = r - q9 * 9;
            int q3 = ry / 3;
            int dy = q3 - 1;
            int dz = ry - q3 * 3 - 1;
            int xx = x + dx, yy = y + dy, zz = z + dz;
            bool ok = (xx >= 0) & (xx < XF) & (yy >= 0) & (yy < YF) & (zz >= 0) & (zz < ZF);
            int xc = xx < 0 ? 0 : (xx > XF - 1 ? XF - 1 : xx);
            int yc = yy < 0 ? 0 : (yy > YF - 1 ? YF - 1 : yy);
            int zc = zz < 0 ? 0 : (zz > ZF - 1 ? ZF - 1 : zz);
            float val = in[(long)ic * NVOX + ((long)xc * YF + yc) * ZF + zc] * (ok ? 1.f : 0.f);
            bm[h] = (_Float16)val;
        }
        v16h am;
        #pragma unroll
        for (int h = 0; h < 16; ++h) {
            int ka = kb + h + ((h >= 8) ? 8 : 0) + (hi ? 8 : 0);
            am[h] = (_Float16)wrow0[ka];
        }
        acc0 = __builtin_amdgcn_wmma_f32_16x16x32_f16(
                   false, am, false, bm, (short)0, acc0, false, false);
        #pragma unroll
        for (int h = 0; h < 16; ++h) {
            int ka = kb + h + ((h >= 8) ? 8 : 0) + (hi ? 8 : 0);
            am[h] = (_Float16)wrow1[ka];
        }
        acc1 = __builtin_amdgcn_wmma_f32_16x16x32_f16(
                   false, am, false, bm, (short)0, acc1, false, false);
    }
    {
        float* p0 = out + (long)(hi ? 8 : 0) * NVOX + n0 + col;
        float* p1 = out + (long)(16 + (hi ? 8 : 0)) * NVOX + n0 + col;
        #pragma unroll
        for (int r = 0; r < 8; ++r) {
            p0[(long)r * NVOX] = acc0[r];
            p1[(long)r * NVOX] = acc1[r];
        }
    }
}

// =====================================================================
// GroupNorm (biased var, eps=1e-5), optional ReLU, in-place.
// blockIdx.x = batch*G + g. gamma/beta indexed by global channel.
// =====================================================================
__global__ __launch_bounds__(256)
void k_groupnorm(float* __restrict__ x, const float* __restrict__ gamma,
                 const float* __restrict__ beta, int C, int G, long S,
                 int relu, long batchStride)
{
    int batch = blockIdx.x / G;
    int g     = blockIdx.x % G;
    int cpg   = C / G;
    float* base = x + (long)batch * batchStride + (long)g * cpg * S;
    long total = (long)cpg * S;
    float s = 0.f, s2 = 0.f;
    for (long i = threadIdx.x; i < total; i += blockDim.x) {
        float vv = base[i]; s += vv; s2 += vv * vv;
    }
    __shared__ float sh[256], sh2[256];
    int t = threadIdx.x;
    sh[t] = s; sh2[t] = s2;
    __syncthreads();
    for (int off = 128; off > 0; off >>= 1) {
        if (t < off) { sh[t] += sh[t + off]; sh2[t] += sh2[t + off]; }
        __syncthreads();
    }
    float mean = sh[0] / (float)total;
    float var  = sh2[0] / (float)total - mean * mean;
    float inv  = rsqrtf(var + 1e-5f);
    for (long i = threadIdx.x; i < total; i += blockDim.x) {
        int c = g * cpg + (int)(i / S);
        float vv = (base[i] - mean) * inv * gamma[c] + beta[c];
        if (relu) vv = fmaxf(vv, 0.f);
        base[i] = vv;
    }
}

// =====================================================================
// Trilinear resize (align_corners=False, half-pixel), separable == 8-tap.
// mode 0: plain, 1: sigmoid epilogue, 2: add residual (same shape as out).
// =====================================================================
__device__ __forceinline__ void axis_coord(int i, int in_size, int out_size,
                                           int& lo, int& hi, float& w)
{
    if (in_size == out_size) { lo = i; hi = i; w = 0.f; return; }
    float scale = (float)in_size / (float)out_size;
    float c = ((float)i + 0.5f) * scale - 0.5f;
    c = fminf(fmaxf(c, 0.f), (float)(in_size - 1));
    lo = (int)floorf(c);
    hi = (lo + 1 < in_size) ? lo + 1 : in_size - 1;
    w  = c - (float)lo;
}

__global__ __launch_bounds__(256)
void k_resize(const float* __restrict__ in, float* __restrict__ out, int C,
              int iX, int iY, int iZ, int oX, int oY, int oZ,
              int mode, const float* __restrict__ add)
{
    long idx = (long)blockIdx.x * blockDim.x + threadIdx.x;
    long ovox = (long)oX * oY * oZ;
    if (idx >= (long)C * ovox) return;
    int  c = (int)(idx / ovox);
    long r = idx - (long)c * ovox;
    int z = (int)(r % oZ);
    int y = (int)((r / oZ) % oY);
    int x = (int)(r / ((long)oZ * oY));
    int x0, x1, y0, y1, z0, z1; float wx, wy, wz;
    axis_coord(x, iX, oX, x0, x1, wx);
    axis_coord(y, iY, oY, y0, y1, wy);
    axis_coord(z, iZ, oZ, z0, z1, wz);
    const float* p = in + (long)c * iX * iY * iZ;
    long sYZ = (long)iY * iZ;
    float v000 = p[x0 * sYZ + (long)y0 * iZ + z0], v001 = p[x0 * sYZ + (long)y0 * iZ + z1];
    float v010 = p[x0 * sYZ + (long)y1 * iZ + z0], v011 = p[x0 * sYZ + (long)y1 * iZ + z1];
    float v100 = p[x1 * sYZ + (long)y0 * iZ + z0], v101 = p[x1 * sYZ + (long)y0 * iZ + z1];
    float v110 = p[x1 * sYZ + (long)y1 * iZ + z0], v111 = p[x1 * sYZ + (long)y1 * iZ + z1];
    float v00 = v000 + (v001 - v000) * wz;
    float v01 = v010 + (v011 - v010) * wz;
    float v10 = v100 + (v101 - v100) * wz;
    float v11 = v110 + (v111 - v110) * wz;
    float v0  = v00 + (v01 - v00) * wy;
    float v1  = v10 + (v11 - v10) * wy;
    float val = v0 + (v1 - v0) * wx;
    if (mode == 1)      val = 1.f / (1.f + __expf(-val));
    else if (mode == 2) val += add[idx];
    out[idx] = val;
}

// =====================================================================
// Six-direction neighbor sampling: generate_grids + grid_sample_3d fused.
// Reproduces torch convention: grid[...,0] indexes the LAST axis (W=ZD).
// =====================================================================
__global__ __launch_bounds__(256)
void k_sample_nb(const float* __restrict__ xd, const float* __restrict__ off,
                 float* __restrict__ nb)
{
    int idx = blockIdx.x * blockDim.x + threadIdx.x;
    if (idx >= 6 * ND) return;
    int n = idx / ND;
    int v = idx - n * ND;
    int z = v % ZD;
    int y = (v / ZD) % YD;
    int x = v / (ZD * YD);
    float o = off[(long)n * ND + v];
    float sx = (float)x, sy = (float)y, sz = (float)z;
    if      (n == 0) sx = fminf(fmaxf(sx + o * (XD - 1), 0.f), (float)(XD - 1));
    else if (n == 1) sx = fminf(fmaxf(sx - o * (XD - 1), 0.f), (float)(XD - 1));
    else if (n == 2) sy = fminf(fmaxf(sy + o * (YD - 1), 0.f), (float)(YD - 1));
    else if (n == 3) sy = fminf(fmaxf(sy - o * (YD - 1), 0.f), (float)(YD - 1));
    else if (n == 4) sz = fminf(fmaxf(sz + o * (ZD - 1), 0.f), (float)(ZD - 1));
    else             sz = fminf(fmaxf(sz - o * (ZD - 1), 0.f), (float)(ZD - 1));
    float g0 = 2.f * sx / (float)(XD - 1) - 1.f;
    float g1 = 2.f * sy / (float)(YD - 1) - 1.f;
    float g2 = 2.f * sz / (float)(ZD - 1) - 1.f;
    // unnormalize per torch: comp0 -> W(=ZD), comp1 -> H(=YD), comp2 -> D(=XD)
    float ix = ((g0 + 1.f) * (float)ZD - 1.f) * 0.5f;
    float iy = ((g1 + 1.f) * (float)YD - 1.f) * 0.5f;
    float iz = ((g2 + 1.f) * (float)XD - 1.f) * 0.5f;
    float fx0 = floorf(ix), fy0 = floorf(iy), fz0 = floorf(iz);
    float wx = ix - fx0, wy = iy - fy0, wz = iz - fz0;
    int x0 = (int)fx0, y0 = (int)fy0, z0 = (int)fz0;
    int   lin[8]; float wgt[8];
    int t = 0;
    #pragma unroll
    for (int dz = 0; dz < 2; ++dz)
    #pragma unroll
    for (int dy = 0; dy < 2; ++dy)
    #pragma unroll
    for (int dx = 0; dx < 2; ++dx) {
        int xc = x0 + dx, yc = y0 + dy, zc = z0 + dz;
        bool ok = (xc >= 0) & (xc < ZD) & (yc >= 0) & (yc < YD) & (zc >= 0) & (zc < XD);
        int xcc = xc < 0 ? 0 : (xc > ZD - 1 ? ZD - 1 : xc);
        int ycc = yc < 0 ? 0 : (yc > YD - 1 ? YD - 1 : yc);
        int zcc = zc < 0 ? 0 : (zc > XD - 1 ? XD - 1 : zc);
        lin[t] = (zcc * YD + ycc) * ZD + xcc;
        wgt[t] = (dx ? wx : 1.f - wx) * (dy ? wy : 1.f - wy) * (dz ? wz : 1.f - wz) * (ok ? 1.f : 0.f);
        ++t;
    }
    for (int c = 0; c < CIN; ++c) {
        const float* p = xd + (long)c * ND;
        float a = 0.f;
        #pragma unroll
        for (int tt = 0; tt < 8; ++tt) a += p[lin[tt]] * wgt[tt];
        nb[((long)n * CIN + c) * ND + v] = a;
    }
}

// =====================================================================
// scores = k.q / sqrt(C); softmax over 6 dirs; agg = sum_n w*v
// =====================================================================
__global__ __launch_bounds__(256)
void k_attn(const float* __restrict__ q, const float* __restrict__ kk,
            const float* __restrict__ vv, float* __restrict__ agg)
{
    int v = blockIdx.x * blockDim.x + threadIdx.x;
    if (v >= ND) return;
    float s[6] = {0.f, 0.f, 0.f, 0.f, 0.f, 0.f};
    for (int c = 0; c < CIN; ++c) {
        float qv = q[(long)c * ND + v];
        #pragma unroll
        for (int n = 0; n < 6; ++n)
            s[n] += kk[((long)n * CIN + c) * ND + v] * qv;
    }
    float mx = s[0] * 0.125f;
    #pragma unroll
    for (int n = 1; n < 6; ++n) mx = fmaxf(mx, s[n] * 0.125f);
    float e[6]; float tot = 0.f;
    #pragma unroll
    for (int n = 0; n < 6; ++n) { e[n] = __expf(s[n] * 0.125f - mx); tot += e[n]; }
    float inv = 1.f / tot;
    #pragma unroll
    for (int n = 0; n < 6; ++n) e[n] *= inv;
    for (int c = 0; c < CIN; ++c) {
        float a = 0.f;
        #pragma unroll
        for (int n = 0; n < 6; ++n)
            a += e[n] * vv[((long)n * CIN + c) * ND + v];
        agg[(long)c * ND + v] = a;
    }
}

// =====================================================================
// Orchestration
// =====================================================================
extern "C" void kernel_launch(void* const* d_in, const int* in_sizes, int n_in,
                              void* d_out, int out_size, void* d_ws, size_t ws_size,
                              hipStream_t stream)
{
    const float* cls_feat  = (const float*)d_in[0];
    const float* box_feat  = (const float*)d_in[1];
    const float* bbox_W1   = (const float*)d_in[2];
    const float* bbox_gn_g = (const float*)d_in[3];
    const float* bbox_gn_b = (const float*)d_in[4];
    const float* bbox_W2   = (const float*)d_in[5];
    const float* cls_W1    = (const float*)d_in[6];
    const float* cls_gn_g  = (const float*)d_in[7];
    const float* cls_gn_b  = (const float*)d_in[8];
    const float* cls_W2    = (const float*)d_in[9];
    const float* rWq       = (const float*)d_in[10];
    const float* rWk       = (const float*)d_in[11];
    const float* rWv       = (const float*)d_in[12];
    const float* rWo       = (const float*)d_in[13];
    const float* rgng      = (const float*)d_in[14];
    const float* rgnb      = (const float*)d_in[15];
    (void)in_sizes; (void)n_in; (void)out_size; (void)ws_size;

    float* out      = (float*)d_out;
    float* cls_out  = out;                     // [18][NVOX]
    float* bbox_out = out + (long)NCLS * NVOX; // [6][NVOX]

    float* ws = (float*)d_ws;
    size_t o = 0;
    float* bufH   = ws + o; o += (size_t)MID * NVOX;       // conv3x3 hidden
    float* P0     = ws + o; o += (size_t)CIN * NVOX;       // upx scratch (aliases nb)
    float* P1     = ws + o; o += (size_t)CIN * NVOX;       // refined x
    float* bufXd  = ws + o; o += (size_t)CIN * ND;
    float* bufQ   = ws + o; o += (size_t)CIN * ND;
    float* bufK   = ws + o; o += (size_t)6 * CIN * ND;
    float* bufV   = ws + o; o += (size_t)6 * CIN * ND;
    float* bufAgg = ws + o; o += (size_t)CIN * ND;
    float* bufOff = ws + o; o += (size_t)6 * ND;
    float* bufNb  = P0;    // nb [6][64][ND] fits in P0; lifetimes disjoint

    const int TB = 256;
    dim3 blk(TB);
    const int GEMM_F = NVOX / 16 / 8;  // 1152 blocks (full res)
    const int GEMM_D = ND / 16 / 8;    // 144 blocks (down res)

    // ---------------- bbox branch ----------------
    k_conv3x3_wmma<<<dim3(GEMM_F), blk, 0, stream>>>(box_feat, bbox_W1, bufH);
    k_groupnorm<<<dim3(32), blk, 0, stream>>>(bufH, bbox_gn_g, bbox_gn_b, MID, 32, (long)NVOX, 1, 0L);
    k_conv1x1_wmma<<<dim3(GEMM_F), blk, 0, stream>>>(bufH, bbox_W2, bbox_out, NBOX, MID, NVOX, 0L, 0L);
    // offsets = sigmoid(downsampled bbox_out)
    k_resize<<<dim3((6 * ND + TB - 1) / TB), blk, 0, stream>>>(
        bbox_out, bufOff, 6, XF, YF, ZF, XD, YD, ZD, 1, nullptr);

    // ---------------- refine stack ----------------
    const float* xin = cls_feat;
    for (int i = 0; i < NLAYER; ++i) {
        const float* Wq = rWq + (long)i * CIN * CIN;
        const float* Wk = rWk + (long)i * CIN * CIN;
        const float* Wv = rWv + (long)i * CIN * CIN;
        const float* Wo = rWo + (long)i * CIN * CIN;
        const float* gg = rgng + (long)i * 4 * CIN;
        const float* gb = rgnb + (long)i * 4 * CIN;

        // downsample x
        k_resize<<<dim3((CIN * ND + TB - 1) / TB), blk, 0, stream>>>(
            xin, bufXd, CIN, XF, YF, ZF, XD, YD, ZD, 0, nullptr);
        // q = GN(Wq * xd)
        k_conv1x1_wmma<<<dim3(GEMM_D), blk, 0, stream>>>(bufXd, Wq, bufQ, CIN, CIN, ND, 0L, 0L);
        k_groupnorm<<<dim3(32), blk, 0, stream>>>(bufQ, gg + 0 * CIN, gb + 0 * CIN, CIN, 32, (long)ND, 0, 0L);
        // neighbors
        k_sample_nb<<<dim3((6 * ND + TB - 1) / TB), blk, 0, stream>>>(bufXd, bufOff, bufNb);
        // k, v = GN(W * nb)  (batched over 6 directions; per-direction GN stats)
        k_conv1x1_wmma<<<dim3(GEMM_D, 1, 6), blk, 0, stream>>>(
            bufNb, Wk, bufK, CIN, CIN, ND, (long)CIN * ND, (long)CIN * ND);
        k_conv1x1_wmma<<<dim3(GEMM_D, 1, 6), blk, 0, stream>>>(
            bufNb, Wv, bufV, CIN, CIN, ND, (long)CIN * ND, (long)CIN * ND);
        k_groupnorm<<<dim3(6 * 32), blk, 0, stream>>>(bufK, gg + 1 * CIN, gb + 1 * CIN, CIN, 32, (long)ND, 0, (long)CIN * ND);
        k_groupnorm<<<dim3(6 * 32), blk, 0, stream>>>(bufV, gg + 2 * CIN, gb + 2 * CIN, CIN, 32, (long)ND, 0, (long)CIN * ND);
        // attention aggregate
        k_attn<<<dim3((ND + TB - 1) / TB), blk, 0, stream>>>(bufQ, bufK, bufV, bufAgg);
        // upsample + residual add -> P0 (nb is dead now)
        k_resize<<<dim3((CIN * NVOX + TB - 1) / TB), blk, 0, stream>>>(
            bufAgg, P0, CIN, XD, YD, ZD, XF, YF, ZF, 2, xin);
        // x_next = GN(Wo * (up + x))
        k_conv1x1_wmma<<<dim3(GEMM_F), blk, 0, stream>>>(P0, Wo, P1, CIN, CIN, NVOX, 0L, 0L);
        k_groupnorm<<<dim3(32), blk, 0, stream>>>(P1, gg + 3 * CIN, gb + 3 * CIN, CIN, 32, (long)NVOX, 0, 0L);
        xin = P1;
    }

    // ---------------- cls branch ----------------
    k_conv3x3_wmma<<<dim3(GEMM_F), blk, 0, stream>>>(xin, cls_W1, bufH);
    k_groupnorm<<<dim3(32), blk, 0, stream>>>(bufH, cls_gn_g, cls_gn_b, MID, 32, (long)NVOX, 1, 0L);
    k_conv1x1_wmma<<<dim3(GEMM_F), blk, 0, stream>>>(bufH, cls_W2, cls_out, NCLS, MID, NVOX, 0L, 0L);
}
